// FourierTemporalAttention_33406255628298
// MI455X (gfx1250) — matrix operand order repaired and checked by reading, not compiled, not run on hardware
//
#include <hip/hip_runtime.h>

typedef __attribute__((ext_vector_type(16))) _Float16 v16h;
typedef __attribute__((ext_vector_type(8)))  _Float16 v8h;
typedef __attribute__((ext_vector_type(8)))  float    v8f;
typedef __attribute__((ext_vector_type(4)))  float    v4f;

#define B_N   128
#define TQ    64
#define TK    200
#define D_N   64
#define DF    64
#define H_N   10
#define KTILE 13          // 13 * 16 = 208 >= 200
#define KPAD  224         // 7 * 32, padded K for context GEMM
#define SEQ_ST 72         // halfs per seq row (144 B, 16B-aligned rows)
#define RAW_ST 112        // halfs per lane raw-score array (224 B)

__global__ __launch_bounds__(64) void fta_kernel(
    const float* __restrict__ seq, const float* __restrict__ dtn,
    const float* __restrict__ target, const int* __restrict__ vmask,
    const float* __restrict__ fr, const float* __restrict__ fi,
    const float* __restrict__ Aw, const float* __restrict__ Ab,
    const float* __restrict__ Aout, float* __restrict__ out)
{
    __shared__ _Float16 sh_seq[KPAD * SEQ_ST];      // seq[k][d] f16, zero-padded k>=200
    __shared__ _Float16 sh_aw[H_N * 64];            // A_w f16
    __shared__ float    sh_decay[208];              // decay * valid (0 if invalid/pad)
    __shared__ float    sh_mask[208];               // 0 or -30000
    __shared__ _Float16 sh_raw[2 * 32 * RAW_ST];    // lane-private raw scores / probs
    __shared__ _Float16 sh_att[2 * 16 * KPAD];      // att[t][k] f16 per wave

    const int tid  = threadIdx.x;
    const int b    = blockIdx.y;
    const int wave = tid >> 5;
    const int lane = tid & 31;
    const int n    = lane & 15;
    const int hi   = lane >> 4;
    const int t0   = (blockIdx.x * 2 + wave) * 16;

    // L2 prefetch of this block's working set (global_prefetch_b8)
    {
        const float* sp = seq + (size_t)b * TK * 64;
        const float* tp = target + (size_t)b * TQ * 64;
        __builtin_prefetch(sp + (size_t)tid * 200, 0, 1);
        __builtin_prefetch(tp + (size_t)tid * 64, 0, 1);
    }

    // ---------------- Phase A: cooperative LDS prep ----------------
    for (int i = tid; i < H_N * 64; i += 64) sh_aw[i] = (_Float16)Aw[i];

    // Fourier decay (symmetric halves collapse: mean_128(...)/2 == sum_64/128)
    for (int k = tid; k < 208; k += 64) {
        float dec = 0.f, msk = -30000.f;
        if (k < TK) {
            float dtv = dtn[b * TK + k];
            float s = 0.f;
            for (int j = 0; j < DF; ++j) {
                float f = ((float)j * (1.0f / 63.0f)) * 0.5f;
                float w = 6.28318530718f * f * dtv;
                s += __cosf(w) * fr[j] - __sinf(w) * fi[j];
            }
            float dv = s * (1.0f / 128.0f);
            dv = fminf(fmaxf(dv, 0.f), 1.f);
            if (vmask[b * TK + k] != 0) { dec = dv; msk = 0.f; }
        }
        sh_decay[k] = dec; sh_mask[k] = msk;
    }

    // seq -> f16 LDS (row-major [k][d]), zero pad rows
    for (int i = tid; i < KPAD * 64; i += 64) {
        int k = i >> 6, dd = i & 63;
        float v = (k < TK) ? seq[(size_t)(b * TK + k) * 64 + dd] : 0.f;
        sh_seq[k * SEQ_ST + dd] = (_Float16)v;
    }
    // zero att (pad rows stay zero)
    for (int i = tid; i < 2 * 16 * KPAD; i += 64) sh_att[i] = (_Float16)0.f;
    __syncthreads();

    // ---------------- per-wave setup ----------------
    float ab[H_N], aow[H_N];
    #pragma unroll
    for (int h = 0; h < H_N; ++h) { ab[h] = Ab[h]; aow[h] = Aout[h]; }

    // target B fragments (K=d rows, N=t cols), kept in registers for whole sweep
    v16h btf[2];
    #pragma unroll
    for (int c = 0; c < 2; ++c) {
        const float* tp = target + (size_t)(b * TQ + t0 + n) * 64 + c * 32 + hi * 8;
        #pragma unroll
        for (int q = 0; q < 4; ++q) {
            v4f f4 = *(const v4f*)(tp + q * 4 + (q >> 1) * 8);
            #pragma unroll
            for (int e = 0; e < 4; ++e) btf[c][q * 4 + e] = (_Float16)f4[e];
        }
    }

    // ---------------- Phase B: score WMMA + flash softmax ----------------
    float m = -1e30f;
    _Float16* rawp = &sh_raw[(wave * 32 + lane) * RAW_ST];

    for (int kt = 0; kt < KTILE; ++kt) {
        v8f acc[H_N];
        #pragma unroll
        for (int h = 0; h < H_N; ++h) acc[h] = v8f{};

        #pragma unroll
        for (int c = 0; c < 2; ++c) {
            const int krow = kt * 16 + n;                 // A layout: M = lane&15
            const _Float16* sp = &sh_seq[krow * SEQ_ST + c * 32 + hi * 8];
            v8h s_lo = *(const v8h*)(sp);
            v8h s_hi = *(const v8h*)(sp + 16);
            // process heads in groups of 5: build 5 A-frags in distinct regs,
            // then fire 5 WMMAs back-to-back (kills WMMA->VALU WAR nops)
            #pragma unroll
            for (int g = 0; g < 2; ++g) {
                v16h afs[5];
                #pragma unroll
                for (int u = 0; u < 5; ++u) {
                    const int h = g * 5 + u;
                    const _Float16* ap = &sh_aw[h * 64 + c * 32 + hi * 8];
                    v8h a_lo = *(const v8h*)(ap);
                    v8h a_hi = *(const v8h*)(ap + 16);
                    v8h p_lo = s_lo * a_lo;               // v_pk_mul_f16
                    v8h p_hi = s_hi * a_hi;
                    #pragma unroll
                    for (int i = 0; i < 8; ++i) { afs[u][i] = p_lo[i]; afs[u][i + 8] = p_hi[i]; }
                }
                #pragma unroll
                for (int u = 0; u < 5; ++u) {
                    const int h = g * 5 + u;
                    acc[h] = __builtin_amdgcn_wmma_f32_16x16x32_f16(
                        false, afs[u], false, btf[c], (short)0, acc[h], false, false);
                }
            }
        }
        // tanh-MLP epilogue: C layout M=k (rows), N=t (lane column)
        #pragma unroll
        for (int e = 0; e < 8; ++e) {
            int kg = kt * 16 + hi * 8 + e;
            float s = 0.f;
            #pragma unroll
            for (int h = 0; h < H_N; ++h) s += aow[h] * tanhf(acc[h][e] + ab[h]);
            float v = s + sh_mask[kg];
            m = fmaxf(m, v);
            rawp[kt * 8 + e] = (_Float16)v;
        }
    }

    // column (t) is split across lane-pair (n, n+16)
    m = fmaxf(m, __shfl_xor(m, 16, 32));
    __syncthreads();

    float ssum = 0.f;
    for (int i = 0; i < 104; i += 8) {
        v8h pv = *(v8h*)(rawp + i);
        v8h ov;
        #pragma unroll
        for (int q = 0; q < 8; ++q) {
            float p = __expf((float)pv[q] - m);
            ssum += p;
            ov[q] = (_Float16)p;
        }
        *(v8h*)(rawp + i) = ov;
    }
    ssum += __shfl_xor(ssum, 16, 32);
    float rn = 1.0f / ssum;
    __syncthreads();

    // normalized att * decay -> sh_att[t][k] (A-operand layout for context GEMM)
    _Float16* attp = &sh_att[(wave * 16 + n) * KPAD];
    for (int i = 0; i < 104; ++i) {
        int kt = i >> 3, e = i & 7;
        int kg = kt * 16 + hi * 8 + e;
        float p = (float)rawp[i] * rn * sh_decay[kg];
        attp[kg] = (_Float16)p;
    }
    __syncthreads();

    // ---------------- Phase C: context = att @ seq via WMMA ----------------
    v8f cacc[4];
    #pragma unroll
    for (int dti = 0; dti < 4; ++dti) cacc[dti] = v8f{};

    const _Float16* awave = &sh_att[wave * 16 * KPAD];
    for (int c = 0; c < 7; ++c) {
        // A frag: att[t][k], M=t=lane&15, K runs c*32+hi*8 (+0..7, +16..23)
        const _Float16* ap = awave + n * KPAD + c * 32 + hi * 8;
        v8h a_lo = *(const v8h*)(ap);
        v8h a_hi = *(const v8h*)(ap + 16);
        v16h afr;
        #pragma unroll
        for (int i = 0; i < 8; ++i) { afr[i] = a_lo[i]; afr[i + 8] = a_hi[i]; }
        #pragma unroll
        for (int dti = 0; dti < 4; ++dti) {
            // B frag: seq[k][d], K=c*32+hi*8 (+0..7, +16..23) rows, N=d=dti*16+n
            v16h bfr;
            #pragma unroll
            for (int kk = 0; kk < 16; ++kk) {
                int kr = c * 32 + hi * 8 + (kk & 7) + (kk >> 3) * 16;           // < 224, zero-padded
                bfr[kk] = sh_seq[kr * SEQ_ST + dti * 16 + n];
            }
            cacc[dti] = __builtin_amdgcn_wmma_f32_16x16x32_f16(
                false, afr, false, bfr, (short)0, cacc[dti], false, false);
        }
    }

    // store: C layout M=t rows, N=d column (lane&15)
    #pragma unroll
    for (int dti = 0; dti < 4; ++dti) {
        #pragma unroll
        for (int r = 0; r < 8; ++r) {
            int t = t0 + hi * 8 + r;
            out[(size_t)(b * TQ + t) * 64 + dti * 16 + n] = cacc[dti][r];
        }
    }
}

extern "C" void kernel_launch(void* const* d_in, const int* in_sizes, int n_in,
                              void* d_out, int out_size, void* d_ws, size_t ws_size,
                              hipStream_t stream) {
    const float* seq    = (const float*)d_in[0];
    const float* dtn    = (const float*)d_in[1];
    const float* target = (const float*)d_in[2];
    const int*   vmask  = (const int*)d_in[3];
    const float* fr     = (const float*)d_in[4];
    const float* fi     = (const float*)d_in[5];
    const float* Aw     = (const float*)d_in[6];
    const float* Ab     = (const float*)d_in[7];
    const float* Aout   = (const float*)d_in[8];

    dim3 grid(2, B_N);   // (t-pair, batch)
    fta_kernel<<<grid, 64, 0, stream>>>(seq, dtn, target, vmask, fr, fi,
                                        Aw, Ab, Aout, (float*)d_out);
}
